// InteractionPredictor_64931315581501
// MI455X (gfx1250) — matrix operand order repaired and checked by reading, not compiled, not run on hardware
//
#include <hip/hip_runtime.h>
#include <hip/hip_bf16.h>
#include <math.h>

// ---------------------------------------------------------------------------
// E(3)-equivariant GNN interaction predictor, fused for gfx1250 (MI455X).
// All big GEMMs ([16 edges x 64] @ [64 x Ncols]) run on v_wmma_f32_16x16x32_f16.
// ---------------------------------------------------------------------------

typedef __attribute__((ext_vector_type(16))) _Float16 v16h;
typedef __attribute__((ext_vector_type(8)))  float    v8f;

#define SQ3F 1.7320508075688772f
#define PW_TP 0.20412414523193154f      // 1/sqrt(24): both scalar & vector path weights
#define PW_REC 0.05590169943749474f     // 1/sqrt(320)

__device__ __forceinline__ float silu_n(float x){
  // e3nn normalize2mom(silu): silu(x) * 1.6789717
  return (x / (1.0f + __expf(-x))) * 1.6789717f;
}

// A fragment (16x32 f16, M=edges): lane m=lane&15; lanes>=16 take K base*2 (=16).
// element s -> K = 2*base + s  => one contiguous 16-half run (matches B).
__device__ __forceinline__ v16h make_afrag(const _Float16* hb, int m, int base, int kofs){
  union { v16h v; float4 q[2]; } u;
  u.q[0] = *(const float4*)(hb + m*64 + kofs + 2*base);
  u.q[1] = *(const float4*)(hb + m*64 + kofs + 2*base + 8);
  return u.v;
}

// C = h[16x64] @ W[64 x ntiles*16]; W pre-transposed f16: WT[col][64] (scale folded).
// B fragment: lane n=lane&15 (column), K contiguous -> one 32B load per K-half.
__device__ __forceinline__ void gemm_to_lds(const _Float16* hb,
                                            const _Float16* __restrict__ wt,
                                            float* wout, int lane, int ntiles, int ldw){
  const int m = lane & 15, g = lane >> 4;
  const int base = g * 8, kb = g * 16, n = m;
  v16h a0 = make_afrag(hb, m, base, 0);
  v16h a1 = make_afrag(hb, m, base, 32);
  for (int t = 0; t < ntiles; ++t){
    const _Float16* wcol = wt + (size_t)(t*16 + n) * 64;
    v16h b0 = *(const v16h*)(wcol + kb);
    v16h b1 = *(const v16h*)(wcol + 32 + kb);
    v8f c = {};
    c = __builtin_amdgcn_wmma_f32_16x16x32_f16(false, a0, false, b0, (short)0, c, false, false);
    c = __builtin_amdgcn_wmma_f32_16x16x32_f16(false, a1, false, b1, (short)0, c, false, false);
#pragma unroll
    for (int r = 0; r < 8; ++r)                 // C: lane n, VGPR r -> row m=r+8g
      wout[(size_t)(r + g*8) * ldw + t*16 + n] = c[r];
  }
}

// ---- prep: WT[col][k] = W1[k][col] * (1/sqrt(64)), f32 -> f16 -----------------
__global__ __launch_bounds__(256) void prep_kernel(const float* __restrict__ src,
                                                   _Float16* __restrict__ dst, int ncols){
  int idx = blockIdx.x * blockDim.x + threadIdx.x;
  if (idx >= ncols * 64) return;
  int col = idx >> 6, k = idx & 63;
  dst[idx] = (_Float16)(src[k * ncols + col] * 0.125f);
}

__global__ __launch_bounds__(256) void zero_kernel(float* __restrict__ p, int n){
  int i = blockIdx.x * blockDim.x + threadIdx.x;
  if (i < n) p[i] = 0.0f;
}

// ---- node embedding: x[N,10] -> 64 (silu) -> 16 scalars, zero geo -------------
__global__ __launch_bounds__(256) void embed_kernel(const float* __restrict__ x,
                                                    const float* __restrict__ w0,
                                                    const float* __restrict__ w1,
                                                    float* __restrict__ nodeOut, int N){
  int n = blockIdx.x * blockDim.x + threadIdx.x;
  if (n >= N) return;
  float in[10];
#pragma unroll
  for (int k = 0; k < 10; ++k) in[k] = x[n*10 + k];
  float h[64];
  for (int j = 0; j < 64; ++j){
    float a = 0.f;
#pragma unroll
    for (int k = 0; k < 10; ++k) a += in[k] * w0[k*64 + j];
    h[j] = silu_n(a * 0.31622776601f);
  }
  for (int o = 0; o < 16; ++o){
    float a = 0.f;
    for (int j = 0; j < 64; ++j) a += h[j] * w1[j*16 + o];
    nodeOut[n*40 + o] = a * 0.125f;
  }
  for (int g = 0; g < 24; ++g) nodeOut[n*40 + 16 + g] = 0.f;
}

// ---- message kernel: one wave32 handles 16 edges ------------------------------
__global__ __launch_bounds__(32) void msg_kernel(const float* __restrict__ nodeIn,
                                                 const float* __restrict__ pos,
                                                 const int*   __restrict__ eidx,
                                                 const float* __restrict__ attr,
                                                 const float* __restrict__ w0,
                                                 const _Float16* __restrict__ wt1,
                                                 float* __restrict__ msgs, int E){
  __shared__ __align__(16) _Float16 hbuf[16*64];
  __shared__ float wbuf[16*576];
  const int lane = threadIdx.x;
  const int tile = blockIdx.x * 16;
  const int el = lane & 15;
  const int e = tile + el;
  const bool valid = e < E;
  const int ec = valid ? e : (E - 1);

  { // hidden: h = silu_n(attr @ w0 / sqrt(5))  -> LDS f16 [16][64]
    float a[5];
#pragma unroll
    for (int k = 0; k < 5; ++k) a[k] = valid ? attr[(size_t)ec*5 + k] : 0.0f;
    const int jb = (lane >> 4) * 32;
    for (int j = 0; j < 32; ++j){
      float acc = 0.f;
#pragma unroll
      for (int k = 0; k < 5; ++k) acc += a[k] * w0[k*64 + jb + j];
      hbuf[el*64 + jb + j] = (_Float16)silu_n(acc * 0.4472135955f);
    }
  }
  __syncthreads();
  gemm_to_lds(hbuf, wt1, wbuf, lane, 36, 576);   // per-edge TP weights [16][576]
  __syncthreads();

  // tensor product (node[src] x sh) with per-edge weights, scatter-add to msgs[dst]
  const int src = eidx[ec];
  const int dst = eidx[E + ec];
  float vx = pos[dst*3+0] - pos[src*3+0];
  float vy = pos[dst*3+1] - pos[src*3+1];
  float vz = pos[dst*3+2] - pos[src*3+2];
  float r = sqrtf(vx*vx + vy*vy + vz*vz);
  float ri = (r > 0.f) ? (SQ3F / r) : 0.f;
  float y1[3] = { vx*ri, vy*ri, vz*ri };

  const float* nf = nodeIn + (size_t)src * 40;
  float xs[16], xv[8][3];
#pragma unroll
  for (int u = 0; u < 16; ++u) xs[u] = nf[u];
#pragma unroll
  for (int u = 0; u < 8; ++u){
    xv[u][0] = nf[16 + u*3 + 0]; xv[u][1] = nf[16 + u*3 + 1]; xv[u][2] = nf[16 + u*3 + 2];
  }
  const float* we = &wbuf[(size_t)el * 576];

  if (lane < 16){           // scalar outputs (16)
    float b[8];
#pragma unroll
    for (int u = 0; u < 8; ++u)
      b[u] = (xv[u][0]*y1[0] + xv[u][1]*y1[1] + xv[u][2]*y1[2]) * (1.0f / SQ3F);
    for (int o = 0; o < 16; ++o){
      float acc = 0.f;
      for (int u = 0; u < 16; ++u) acc += xs[u] * we[u*16 + o];
      for (int u = 0; u < 8;  ++u) acc += b[u]  * we[256 + u*16 + o];
      if (valid)
        __hip_atomic_fetch_add(&msgs[(size_t)dst*40 + o], PW_TP * acc,
                               __ATOMIC_RELAXED, __HIP_MEMORY_SCOPE_AGENT);
    }
  } else {                  // vector outputs (8 x 3)
    for (int o = 0; o < 8; ++o){
      float t = 0.f;
      for (int u = 0; u < 16; ++u) t += xs[u] * we[384 + u*8 + o];
#pragma unroll
      for (int i = 0; i < 3; ++i){
        float acc = t * y1[i];
        for (int u = 0; u < 8; ++u) acc += xv[u][i] * we[512 + u*8 + o];
        if (valid)
          __hip_atomic_fetch_add(&msgs[(size_t)dst*40 + 16 + o*3 + i], PW_TP * acc,
                                 __ATOMIC_RELAXED, __HIP_MEMORY_SCOPE_AGENT);
      }
    }
  }
}

// ---- node update --------------------------------------------------------------
__global__ __launch_bounds__(256) void update_kernel(const float* __restrict__ nodeIn,
                                                     const float* __restrict__ msgs,
                                                     const float* __restrict__ imp,
                                                     const float* __restrict__ uw0,
                                                     const float* __restrict__ uw1,
                                                     float* __restrict__ nodeOut,
                                                     int N, float invSqrtDeg, float geoScale){
  int n = blockIdx.x * blockDim.x + threadIdx.x;
  if (n >= N) return;
  float sc = imp[0] * invSqrtDeg;
  float in[32];
#pragma unroll
  for (int k = 0; k < 16; ++k) in[k]      = msgs[(size_t)n*40 + k] * sc;
#pragma unroll
  for (int k = 0; k < 16; ++k) in[16 + k] = nodeIn[(size_t)n*40 + k];
  float h[64];
  for (int j = 0; j < 64; ++j){
    float a = 0.f;
    for (int k = 0; k < 32; ++k) a += in[k] * uw0[k*64 + j];
    h[j] = silu_n(a * 0.17677669529f);
  }
  for (int o = 0; o < 16; ++o){
    float a = 0.f;
    for (int j = 0; j < 64; ++j) a += h[j] * uw1[j*16 + o];
    nodeOut[(size_t)n*40 + o] = a * 0.125f;
  }
  for (int g = 0; g < 24; ++g)
    nodeOut[(size_t)n*40 + 16 + g] =
      (msgs[(size_t)n*40 + 16 + g] * sc + nodeIn[(size_t)n*40 + 16 + g]) * geoScale;
}

// ---- final inter-edge kernel: one wave32 handles 16 inter-edges ---------------
__global__ __launch_bounds__(32) void inter_kernel(const float* __restrict__ node,
                                                   const float* __restrict__ pos,
                                                   const int*   __restrict__ iidx,
                                                   const float* __restrict__ ligw0,
                                                   const _Float16* __restrict__ ligwt1,
                                                   const float* __restrict__ recw0,
                                                   const _Float16* __restrict__ recwt1,
                                                   float* __restrict__ out, int EI){
  __shared__ __align__(16) _Float16 hL[16*64];
  __shared__ __align__(16) _Float16 hR[16*64];
  __shared__ union { float wbuf[16*576]; float coeff[16*320]; } wc;  // aliased phases
  __shared__ float demb[16*20];
  __shared__ float le[16*40];
  __shared__ float nr[16*40];
  __shared__ float outb[16*8];

  const int lane = threadIdx.x;
  const int tile = blockIdx.x * 16;
  const int el = lane & 15;
  const int e = tile + el;
  const bool valid = e < EI;
  const int ec = valid ? e : (EI - 1);
  const int rec = iidx[ec];
  const int lig = iidx[EI + ec];

  float vx = pos[lig*3+0] - pos[rec*3+0];
  float vy = pos[lig*3+1] - pos[rec*3+1];
  float vz = pos[lig*3+2] - pos[rec*3+2];
  float dist = sqrtf(vx*vx + vy*vy + vz*vz);
  float ri = (dist > 0.f) ? (SQ3F / dist) : 0.f;
  float y1[3] = { vx*ri, vy*ri, vz*ri };

  { // smooth_finite distance basis * sqrt(20)
    const float DC = 1.14136f * 7.38905610f * 4.47213595f;
    for (int k = 0; k < 10; ++k){
      int j = (lane >> 4) * 10 + k;
      float diff = dist * 4.2f - (float)(j + 1);      // (d - v_j)/step
      float sp = diff + 1.0f, sm = 1.0f - diff;
      float a  = (sp > 0.f) ? __expf(-1.0f / sp) : 0.f;
      float b2 = (sm > 0.f) ? __expf(-1.0f / sm) : 0.f;
      demb[el*20 + j] = valid ? (DC * a * b2) : 0.f;
    }
  }
  __syncthreads();

  { // hidden layers for lig (576-w) and rec (2560-w) MLPs
    const int jb = (lane >> 4) * 32;
    for (int j = 0; j < 32; ++j){
      float aL = 0.f, aR = 0.f;
      for (int k = 0; k < 20; ++k){
        float d = demb[el*20 + k];
        aL += d * ligw0[k*64 + jb + j];
        aR += d * recw0[k*64 + jb + j];
      }
      hL[el*64 + jb + j] = (_Float16)silu_n(aL * 0.2236067977f);
      hR[el*64 + jb + j] = (_Float16)silu_n(aR * 0.2236067977f);
    }
  }
  __syncthreads();
  gemm_to_lds(hL, ligwt1, wc.wbuf, lane, 36, 576);   // lig TP weights
  __syncthreads();

  { // lig_emb = TP(node[lig], sh; w)  -> le[16][40]; also stage node[rec] -> nr
    const float* nf = node + (size_t)lig * 40;
    float xs[16], xv[8][3];
#pragma unroll
    for (int u = 0; u < 16; ++u) xs[u] = nf[u];
#pragma unroll
    for (int u = 0; u < 8; ++u){
      xv[u][0] = nf[16+u*3+0]; xv[u][1] = nf[16+u*3+1]; xv[u][2] = nf[16+u*3+2];
    }
    const float* we = &wc.wbuf[(size_t)el * 576];
    if (lane < 16){
      float b[8];
#pragma unroll
      for (int u = 0; u < 8; ++u)
        b[u] = (xv[u][0]*y1[0] + xv[u][1]*y1[1] + xv[u][2]*y1[2]) * (1.0f / SQ3F);
      for (int o = 0; o < 16; ++o){
        float acc = 0.f;
        for (int u = 0; u < 16; ++u) acc += xs[u] * we[u*16 + o];
        for (int u = 0; u < 8;  ++u) acc += b[u]  * we[256 + u*16 + o];
        le[el*40 + o] = PW_TP * acc;
      }
      for (int o = 0; o < 8; ++o) outb[el*8 + o] = 0.f;   // zero result buffer
    } else {
      for (int o = 0; o < 8; ++o){
        float t = 0.f;
        for (int u = 0; u < 16; ++u) t += xs[u] * we[384 + u*8 + o];
#pragma unroll
        for (int i = 0; i < 3; ++i){
          float acc = t * y1[i];
          for (int u = 0; u < 8; ++u) acc += xv[u][i] * we[512 + u*8 + o];
          le[el*40 + 16 + o*3 + i] = PW_TP * acc;
        }
      }
    }
    for (int idx = lane; idx < 16*40; idx += 32){
      int ee = tile + idx / 40, ii = idx % 40;
      int rr = iidx[(ee < EI) ? ee : (EI - 1)];
      nr[idx] = node[(size_t)rr*40 + ii];
    }
  }
  __syncthreads();

  { // bilinear coefficients for tp_rec: 256 (xs_u*ys_v) + 64 (xv_u . yv_v / sqrt3)
    for (int idx = lane; idx < 16*320; idx += 32){
      int m = idx / 320, p = idx % 320;
      float cv;
      if (p < 256){
        int u = p >> 4, vv = p & 15;
        cv = PW_REC * le[m*40 + u] * nr[m*40 + vv];
      } else {
        int q = p - 256, u = q >> 3, vv = q & 7;
        float ds = le[m*40+16+u*3+0]*nr[m*40+16+vv*3+0]
                 + le[m*40+16+u*3+1]*nr[m*40+16+vv*3+1]
                 + le[m*40+16+u*3+2]*nr[m*40+16+vv*3+2];
        cv = (PW_REC / SQ3F) * ds;
      }
      wc.coeff[idx] = cv;     // overwrites wbuf: safe, wbuf fully consumed above
    }
  }
  __syncthreads();

  { // rec GEMM (160 tiles) consumed in-register: out[e][o] += coeff[e][p]*w[e][p*8+o]
    const int m = lane & 15, g = lane >> 4;
    const int base = g*8, kb = g*16, n = m;
    const int o = n & 7, ph = n >> 3;            // tile-invariant column role
    v16h a0 = make_afrag(hR, m, base, 0);
    v16h a1 = make_afrag(hR, m, base, 32);
    float acc[8] = {0.f,0.f,0.f,0.f,0.f,0.f,0.f,0.f};
    for (int t = 0; t < 160; ++t){
      const _Float16* wcol = recwt1 + (size_t)(t*16 + n) * 64;
      if (t + 1 < 160) __builtin_prefetch(recwt1 + (size_t)((t+1)*16 + n) * 64, 0, 1);
      v16h b0 = *(const v16h*)(wcol + kb);
      v16h b1 = *(const v16h*)(wcol + 32 + kb);
      v8f c = {};
      c = __builtin_amdgcn_wmma_f32_16x16x32_f16(false, a0, false, b0, (short)0, c, false, false);
      c = __builtin_amdgcn_wmma_f32_16x16x32_f16(false, a1, false, b1, (short)0, c, false, false);
      int p = 2*t + ph;
#pragma unroll
      for (int r = 0; r < 8; ++r)
        acc[r] += wc.coeff[(r + g*8)*320 + p] * c[r];
    }
#pragma unroll
    for (int r = 0; r < 8; ++r)
      __hip_atomic_fetch_add(&outb[(r + g*8)*8 + o], acc[r],
                             __ATOMIC_RELAXED, __HIP_MEMORY_SCOPE_WORKGROUP);
  }
  __syncthreads();

  if (lane < 16 && (tile + lane) < EI){
#pragma unroll
    for (int o = 0; o < 8; ++o)
      out[(size_t)(tile + lane)*8 + o] = outb[lane*8 + o];
  }
}

// ---------------------------------------------------------------------------
extern "C" void kernel_launch(void* const* d_in, const int* in_sizes, int n_in,
                              void* d_out, int out_size, void* d_ws, size_t ws_size,
                              hipStream_t stream){
  const float* x        = (const float*)d_in[0];
  const float* pos      = (const float*)d_in[1];
  const int*   eidx     = (const int*)  d_in[2];
  const float* eattr    = (const float*)d_in[3];
  const int*   iidx     = (const int*)  d_in[4];
  const float* emb_w0   = (const float*)d_in[5];
  const float* emb_w1   = (const float*)d_in[6];
  const float* imp0     = (const float*)d_in[7];
  const float* msg0_w0  = (const float*)d_in[8];
  const float* msg0_w1  = (const float*)d_in[9];
  const float* upd0_w0  = (const float*)d_in[10];
  const float* upd0_w1  = (const float*)d_in[11];
  const float* imp1     = (const float*)d_in[12];
  const float* msg1_w0  = (const float*)d_in[13];
  const float* msg1_w1  = (const float*)d_in[14];
  const float* upd1_w0  = (const float*)d_in[15];
  const float* upd1_w1  = (const float*)d_in[16];
  const float* lig_w0   = (const float*)d_in[17];
  const float* lig_w1   = (const float*)d_in[18];
  const float* rec_w0   = (const float*)d_in[19];
  const float* rec_w1   = (const float*)d_in[20];

  const int N  = in_sizes[0] / 10;
  const int E  = in_sizes[3] / 5;
  const int EI = in_sizes[4] / 2;

  // workspace carve-up
  size_t off = 0;
  auto carve = [&](size_t bytes) -> void* {
    void* p = (char*)d_ws + off;
    off += (bytes + 255) & ~(size_t)255;
    return p;
  };
  _Float16* wtMsg0 = (_Float16*)carve((size_t)576*64*2);
  _Float16* wtMsg1 = (_Float16*)carve((size_t)576*64*2);
  _Float16* wtLig  = (_Float16*)carve((size_t)576*64*2);
  _Float16* wtRec  = (_Float16*)carve((size_t)2560*64*2);
  float* nodeA = (float*)carve((size_t)N*40*4);
  float* nodeB = (float*)carve((size_t)N*40*4);
  float* msgs  = (float*)carve((size_t)N*40*4);

  const float invSqrtDeg = sqrtf((float)N / (float)E);
  dim3 b256(256);

  // weight prep (f32 -> transposed, scaled f16)
  prep_kernel<<<(576*64 + 255)/256,  b256, 0, stream>>>(msg0_w1, wtMsg0, 576);
  prep_kernel<<<(576*64 + 255)/256,  b256, 0, stream>>>(msg1_w1, wtMsg1, 576);
  prep_kernel<<<(576*64 + 255)/256,  b256, 0, stream>>>(lig_w1,  wtLig,  576);
  prep_kernel<<<(2560*64 + 255)/256, b256, 0, stream>>>(rec_w1,  wtRec,  2560);

  // node embedding
  embed_kernel<<<(N + 255)/256, b256, 0, stream>>>(x, emb_w0, emb_w1, nodeA, N);

  // MP step 0
  zero_kernel<<<(N*40 + 255)/256, b256, 0, stream>>>(msgs, N*40);
  msg_kernel<<<(E + 15)/16, dim3(32), 0, stream>>>(nodeA, pos, eidx, eattr,
                                                   msg0_w0, wtMsg0, msgs, E);
  update_kernel<<<(N + 255)/256, b256, 0, stream>>>(nodeA, msgs, imp0, upd0_w0, upd0_w1,
                                                    nodeB, N, invSqrtDeg, 1.0f);
  // MP step 1
  zero_kernel<<<(N*40 + 255)/256, b256, 0, stream>>>(msgs, N*40);
  msg_kernel<<<(E + 15)/16, dim3(32), 0, stream>>>(nodeB, pos, eidx, eattr,
                                                   msg1_w0, wtMsg1, msgs, E);
  update_kernel<<<(N + 255)/256, b256, 0, stream>>>(nodeB, msgs, imp1, upd1_w0, upd1_w1,
                                                    nodeA, N, invSqrtDeg, 0.5f);

  // final inter-edge head
  inter_kernel<<<(EI + 15)/16, dim3(32), 0, stream>>>(nodeA, pos, iidx,
                                                      lig_w0, wtLig, rec_w0, wtRec,
                                                      (float*)d_out, EI);
}
